// ODEFunc_80496277062094
// MI455X (gfx1250) — compile-verified, matrix operand order checked by reading, not run on hardware
//
#include <hip/hip_runtime.h>
#include <hip/hip_bf16.h>
#include <math.h>

typedef __attribute__((ext_vector_type(16))) __bf16 v16bf;
typedef __attribute__((ext_vector_type(8)))  float  v8f;

namespace {
constexpr int B_  = 4096;
constexpr int T_  = 100;
constexpr int XS_ = 32;
constexpr int AS_ = 32;
constexpr int XH_ = 256;
constexpr int AH_ = 256;
constexpr float DTC = 0.05f;

constexpr int MT       = 32;   // batch rows per workgroup (2 WMMA M-tiles)
constexpr int NTHREADS = 512;  // 16 wave32s
constexpr int NWAVES   = 16;

// bf16 weight pack offsets inside d_ws (in elements)
constexpr size_t OFF_W_IH_A = 0;                          // [768, 64]
constexpr size_t OFF_W_HH_A = OFF_W_IH_A + 768ull * 64;   // [768, 256]
constexpr size_t OFF_W_IH_X = OFF_W_HH_A + 768ull * 256;  // [768, 32]
constexpr size_t OFF_W_HH_X = OFF_W_IH_X + 768ull * 32;   // [768, 256]
constexpr size_t OFF_OA_W1  = OFF_W_HH_X + 768ull * 256;  // [256, 256]
constexpr size_t OFF_OA_W2  = OFF_OA_W1  + 256ull * 256;  // [256, 256]
constexpr size_t OFF_OX_W1  = OFF_OA_W2  + 256ull * 256;  // [256, 256]
constexpr size_t OFF_OX_W2  = OFF_OX_W1  + 256ull * 256;  // [256, 256]
constexpr size_t OFF_DEC_W1 = OFF_OX_W2  + 256ull * 256;  // [256, 512]
constexpr size_t OFF_DEC_W2 = OFF_DEC_W1 + 256ull * 512;  // [ 32, 256]
constexpr size_t WS_BF16_ELEMS = OFF_DEC_W2 + 32ull * 256;
} // namespace

__device__ __forceinline__ float sigm_(float v) { return 1.0f / (1.0f + expf(-v)); }

// C[32,N] (row-major, LDS, f32) = A[32,K] (row-major LDS bf16) @ W[N,K]^T (global bf16) + bias
// Each wave owns 16x16 N-tiles round-robin; per tile it keeps TWO f32 accumulators
// (M rows 0-15 and 16-31) so every B fragment fetched from L2 feeds 2 WMMAs.
template <int N, int K>
__device__ __forceinline__ void block_gemm(const __bf16* __restrict__ As,
                                           const __bf16* __restrict__ W,
                                           const float* __restrict__ bias,
                                           float* __restrict__ C,
                                           int wave, int lane) {
  const int row = lane & 15;
  const int kh  = (lane >> 4) * 16;
  for (int nt = wave; nt < N / 16; nt += NWAVES) {
    const int n0 = nt * 16;
    v8f acc0 = {};
    v8f acc1 = {};
#pragma unroll
    for (int k0 = 0; k0 < K; k0 += 32) {
      v16bf b  = *(const v16bf*)(W + (size_t)(n0 + row) * K + k0 + kh);
      v16bf a0 = *(const v16bf*)(As + row * K + k0 + kh);
      v16bf a1 = *(const v16bf*)(As + (16 + row) * K + k0 + kh);
      acc0 = __builtin_amdgcn_wmma_f32_16x16x32_bf16(false, a0, false, b,
                                                     (short)0, acc0, false, false);
      acc1 = __builtin_amdgcn_wmma_f32_16x16x32_bf16(false, a1, false, b,
                                                     (short)0, acc1, false, false);
    }
    // D layout: lane L, reg r -> (M = r + 8*(L>=16), Ncol = n0 + (L&15))
    const int mrow = (lane >> 4) * 8;
    const int ncol = n0 + (lane & 15);
    const float bv = bias[ncol];
#pragma unroll
    for (int r = 0; r < 8; ++r) {
      C[(mrow + r) * N + ncol]      = acc0[r] + bv;
      C[(16 + mrow + r) * N + ncol] = acc1[r] + bv;
    }
  }
}

// RK4 over h[32,256]: f(u) = tanh(u@W1^T+b1)@W2^T+b2 ; h += dt/6*(k1+2k2+2k3+k4)
__device__ void rk4_block(float* __restrict__ h,
                          const __bf16* __restrict__ W1, const float* __restrict__ b1,
                          const __bf16* __restrict__ W2, const float* __restrict__ b2,
                          float dt, __bf16* __restrict__ abf,
                          float* __restrict__ kacc, float* __restrict__ kcur,
                          float* __restrict__ t1, int tid, int wave, int lane) {
  const float coef[4] = {0.0f, 0.5f, 0.5f, 1.0f};
  const float wgt[4]  = {1.0f, 2.0f, 2.0f, 1.0f};
#pragma unroll 1
  for (int s = 0; s < 4; ++s) {
    const float alpha = coef[s] * dt;
    for (int i = tid; i < MT * 256; i += NTHREADS) {
      float u = (s == 0) ? h[i] : (h[i] + alpha * kcur[i]);
      abf[i] = (__bf16)u;
    }
    __syncthreads();
    block_gemm<256, 256>(abf, W1, b1, t1, wave, lane);   // t1 = u@W1^T + b1
    __syncthreads();
    for (int i = tid; i < MT * 256; i += NTHREADS)
      abf[i] = (__bf16)tanhf(t1[i]);
    __syncthreads();
    block_gemm<256, 256>(abf, W2, b2, kcur, wave, lane); // kcur = tanh(.)@W2^T + b2
    __syncthreads();
    for (int i = tid; i < MT * 256; i += NTHREADS)
      kacc[i] = (s == 0 ? 0.0f : kacc[i]) + wgt[s] * kcur[i];
    __syncthreads();
  }
  const float sc = dt * (1.0f / 6.0f);
  for (int i = tid; i < MT * 256; i += NTHREADS)
    h[i] += sc * kacc[i];
  __syncthreads();
}

__global__ void cvt_bf16_kernel(const float* __restrict__ src, __bf16* __restrict__ dst, int n) {
  int i = blockIdx.x * 256 + threadIdx.x;
  if (i < n) dst[i] = (__bf16)src[i];
}

__global__ void __launch_bounds__(NTHREADS)
node_scan_kernel(const float* __restrict__ x, const float* __restrict__ y,
                 const float* __restrict__ ts, const __bf16* __restrict__ wbf,
                 const float* __restrict__ b_ih_a, const float* __restrict__ b_hh_a,
                 const float* __restrict__ b_ih_x, const float* __restrict__ b_hh_x,
                 const float* __restrict__ oa_b1, const float* __restrict__ oa_b2,
                 const float* __restrict__ ox_b1, const float* __restrict__ ox_b2,
                 const float* __restrict__ dec_b1, const float* __restrict__ dec_b2,
                 float* __restrict__ x_out, float* __restrict__ z_a_out) {
  __shared__ float  s_zx[MT * XH_];
  __shared__ float  s_za[MT * AH_];
  __shared__ float  s_xc[MT * XS_];
  __shared__ float  s_at[MT * AS_];
  __shared__ float  s_mask[MT];
  __shared__ float  s_g1[MT * 768];   // gi / kacc,kcur / dec hidden
  __shared__ float  s_g2[MT * 768];   // gh / t1 / dec out
  __shared__ __bf16 s_abf[MT * 512];  // bf16 A-operand staging

  const int tid  = threadIdx.x;
  const int lane = tid & 31;
  const int wave = tid >> 5;
  const int b0   = blockIdx.x * MT;

  // ---- init: xc = x[:,0,:], z_x = z_a = 0 ----
  for (int i = tid; i < MT * XS_; i += NTHREADS) {
    int m = i >> 5, c = i & 31;
    s_xc[i] = x[(size_t)(b0 + m) * T_ * XS_ + c];
  }
  for (int i = tid; i < MT * 256; i += NTHREADS) { s_zx[i] = 0.0f; s_za[i] = 0.0f; }
  __syncthreads();

  // ---- t = 0 outputs ----
  for (int i = tid; i < MT * XS_; i += NTHREADS) {
    int m = i >> 5, c = i & 31;
    x_out[(size_t)(b0 + m) * T_ * XS_ + c] = s_xc[i];
  }
  for (int i = tid; i < MT * AH_; i += NTHREADS) {
    int m = i >> 8, c = i & 255;
    z_a_out[(size_t)(b0 + m) * AH_ + c] = 0.0f;   // t=0 plane
  }

#pragma unroll 1
  for (int step = 0; step < T_ - 1; ++step) {
    // ---- a_t = y[:, step, :], observation mask ----
    for (int i = tid; i < MT * AS_; i += NTHREADS) {
      int m = i >> 5, c = i & 31;
      s_at[i] = y[(size_t)(b0 + m) * T_ * AS_ + (size_t)step * AS_ + c];
    }
    __syncthreads();
    if (tid < MT) {
      float ssum = 0.0f;
      for (int c = 0; c < AS_; ++c) ssum += s_at[tid * AS_ + c];
      s_mask[tid] = (ssum != 0.0f) ? 1.0f : 0.0f;
    }
    __syncthreads();

    // ---- ODE evolution (skip first step) ----
    if (step > 0) {
      const float dt = ts[step] - ts[step - 1];
      rk4_block(s_za, wbf + OFF_OA_W1, oa_b1, wbf + OFF_OA_W2, oa_b2, dt,
                s_abf, s_g1, s_g1 + MT * 256, s_g2, tid, wave, lane);
      rk4_block(s_zx, wbf + OFF_OX_W1, ox_b1, wbf + OFF_OX_W2, ox_b2, dt,
                s_abf, s_g1, s_g1 + MT * 256, s_g2, tid, wave, lane);
    }

    // ---- GRU(a): inp = [xc, a_t] (K=64) ----
    for (int i = tid; i < MT * 64; i += NTHREADS) {
      int m = i >> 6, k = i & 63;
      float v = (k < 32) ? s_xc[m * 32 + k] : s_at[m * 32 + (k - 32)];
      s_abf[i] = (__bf16)v;
    }
    __syncthreads();
    block_gemm<768, 64>(s_abf, wbf + OFF_W_IH_A, b_ih_a, s_g1, wave, lane);
    __syncthreads();
    for (int i = tid; i < MT * AH_; i += NTHREADS) s_abf[i] = (__bf16)s_za[i];
    __syncthreads();
    block_gemm<768, 256>(s_abf, wbf + OFF_W_HH_A, b_hh_a, s_g2, wave, lane);
    __syncthreads();
    for (int i = tid; i < MT * AH_; i += NTHREADS) {
      int m = i >> 8, c = i & 255;
      const float* gi = s_g1 + m * 768;
      const float* gh = s_g2 + m * 768;
      float r  = sigm_(gi[c]       + gh[c]);
      float zg = sigm_(gi[256 + c] + gh[256 + c]);
      float n  = tanhf(gi[512 + c] + r * gh[512 + c]);
      float hn = (1.0f - zg) * n + zg * s_za[i];
      float mk = s_mask[m];
      s_za[i] = mk * hn + (1.0f - mk) * s_za[i];
    }
    __syncthreads();

    // ---- GRU(x): inp = xc (K=32) ----
    for (int i = tid; i < MT * XS_; i += NTHREADS) s_abf[i] = (__bf16)s_xc[i];
    __syncthreads();
    block_gemm<768, 32>(s_abf, wbf + OFF_W_IH_X, b_ih_x, s_g1, wave, lane);
    __syncthreads();
    for (int i = tid; i < MT * XH_; i += NTHREADS) s_abf[i] = (__bf16)s_zx[i];
    __syncthreads();
    block_gemm<768, 256>(s_abf, wbf + OFF_W_HH_X, b_hh_x, s_g2, wave, lane);
    __syncthreads();
    for (int i = tid; i < MT * XH_; i += NTHREADS) {
      int m = i >> 8, c = i & 255;
      const float* gi = s_g1 + m * 768;
      const float* gh = s_g2 + m * 768;
      float r  = sigm_(gi[c]       + gh[c]);
      float zg = sigm_(gi[256 + c] + gh[256 + c]);
      float n  = tanhf(gi[512 + c] + r * gh[512 + c]);
      s_zx[i] = (1.0f - zg) * n + zg * s_zx[i];
    }
    __syncthreads();

    // ---- decoder: z = [z_x, z_a] (K=512) -> leaky_relu -> [32,32] ----
    for (int i = tid; i < MT * 512; i += NTHREADS) {
      int m = i >> 9, k = i & 511;
      float v = (k < 256) ? s_zx[m * 256 + k] : s_za[m * 256 + (k - 256)];
      s_abf[i] = (__bf16)v;
    }
    __syncthreads();
    block_gemm<256, 512>(s_abf, wbf + OFF_DEC_W1, dec_b1, s_g1, wave, lane);
    __syncthreads();
    for (int i = tid; i < MT * 256; i += NTHREADS) {
      float v = s_g1[i];
      s_abf[i] = (__bf16)(v > 0.0f ? v : 0.01f * v);
    }
    __syncthreads();
    block_gemm<32, 256>(s_abf, wbf + OFF_DEC_W2, dec_b2, s_g2, wave, lane);
    __syncthreads();

    // ---- x update + outputs for t = step+1 ----
    for (int i = tid; i < MT * XS_; i += NTHREADS) {
      int m = i >> 5, c = i & 31;
      float xn = s_xc[i] + DTC * s_g2[m * 32 + c];
      s_xc[i] = xn;
      x_out[(size_t)(b0 + m) * T_ * XS_ + (size_t)(step + 1) * XS_ + c] = xn;
    }
    for (int i = tid; i < MT * AH_; i += NTHREADS) {
      int m = i >> 8, c = i & 255;
      z_a_out[(size_t)(step + 1) * B_ * AH_ + (size_t)(b0 + m) * AH_ + c] = s_za[i];
    }
    __syncthreads();
  }
}

extern "C" void kernel_launch(void* const* d_in, const int* in_sizes, int n_in,
                              void* d_out, int out_size, void* d_ws, size_t ws_size,
                              hipStream_t stream) {
  (void)in_sizes; (void)n_in; (void)out_size; (void)ws_size;

  const float* x      = (const float*)d_in[0];
  const float* y      = (const float*)d_in[1];
  const float* ts     = (const float*)d_in[2];
  const float* W_ih_a = (const float*)d_in[3];
  const float* W_hh_a = (const float*)d_in[4];
  const float* b_ih_a = (const float*)d_in[5];
  const float* b_hh_a = (const float*)d_in[6];
  const float* W_ih_x = (const float*)d_in[7];
  const float* W_hh_x = (const float*)d_in[8];
  const float* b_ih_x = (const float*)d_in[9];
  const float* b_hh_x = (const float*)d_in[10];
  const float* oa_W1  = (const float*)d_in[11];
  const float* oa_b1  = (const float*)d_in[12];
  const float* oa_W2  = (const float*)d_in[13];
  const float* oa_b2  = (const float*)d_in[14];
  const float* ox_W1  = (const float*)d_in[15];
  const float* ox_b1  = (const float*)d_in[16];
  const float* ox_W2  = (const float*)d_in[17];
  const float* ox_b2  = (const float*)d_in[18];
  const float* dec_W1 = (const float*)d_in[19];
  const float* dec_b1 = (const float*)d_in[20];
  const float* dec_W2 = (const float*)d_in[21];
  const float* dec_b2 = (const float*)d_in[22];

  __bf16* wbf = (__bf16*)d_ws;
  auto cvt = [&](const float* src, size_t off, int n) {
    cvt_bf16_kernel<<<(n + 255) / 256, 256, 0, stream>>>(src, wbf + off, n);
  };
  cvt(W_ih_a, OFF_W_IH_A, 768 * 64);
  cvt(W_hh_a, OFF_W_HH_A, 768 * 256);
  cvt(W_ih_x, OFF_W_IH_X, 768 * 32);
  cvt(W_hh_x, OFF_W_HH_X, 768 * 256);
  cvt(oa_W1,  OFF_OA_W1,  256 * 256);
  cvt(oa_W2,  OFF_OA_W2,  256 * 256);
  cvt(ox_W1,  OFF_OX_W1,  256 * 256);
  cvt(ox_W2,  OFF_OX_W2,  256 * 256);
  cvt(dec_W1, OFF_DEC_W1, 256 * 512);
  cvt(dec_W2, OFF_DEC_W2, 32 * 256);

  float* x_out   = (float*)d_out;                           // (B, T, XS)
  float* z_a_out = x_out + (size_t)B_ * T_ * XS_;           // (T, B, AH)

  node_scan_kernel<<<B_ / MT, NTHREADS, 0, stream>>>(
      x, y, ts, (const __bf16*)wbf,
      b_ih_a, b_hh_a, b_ih_x, b_hh_x,
      oa_b1, oa_b2, ox_b1, ox_b2, dec_b1, dec_b2,
      x_out, z_a_out);
}
